// FeedForward_75153337745431
// MI455X (gfx1250) — 8-bit matrix operand order repaired and checked by reading, compiled, not run on hardware
//
#include <hip/hip_runtime.h>
#include <hip/hip_bf16.h>

typedef __attribute__((ext_vector_type(8))) int v8i;

#define EPSF 1e-5f

static constexpr int DIM   = 1024;
static constexpr int HID   = 4096;
static constexpr int MROWS = 8 * 2048;

static constexpr size_t OFF_STATS = 0;
static constexpr size_t OFF_XQ    = 256;
static constexpr size_t OFF_SX    = OFF_XQ  + (size_t)MROWS * DIM;
static constexpr size_t OFF_W1Q   = OFF_SX  + (size_t)MROWS * 4;
static constexpr size_t OFF_W2Q   = OFF_W1Q + (size_t)HID * DIM;
static constexpr size_t OFF_H     = OFF_W2Q + (size_t)DIM * HID;
static constexpr size_t OFF_HQ    = OFF_H   + (size_t)MROWS * HID * 4;
static constexpr size_t OFF_SH    = OFF_HQ  + (size_t)MROWS * HID;

#if __has_builtin(__builtin_amdgcn_global_load_async_to_lds_b128) && \
    __has_builtin(__builtin_amdgcn_s_wait_asynccnt)
#define USE_ASYNC_LDS 1
typedef int vs4i __attribute__((vector_size(4 * sizeof(int))));
#define GLOBAL_AS1(p) ((__attribute__((address_space(1))) vs4i*)(unsigned long long)(p))
#define LDS_AS3(p)    ((__attribute__((address_space(3))) vs4i*)(unsigned)(unsigned long long)(p))
__device__ __forceinline__ void async_copy16(const void* g, void* l) {
    __builtin_amdgcn_global_load_async_to_lds_b128(GLOBAL_AS1(g), LDS_AS3(l), 0, 0);
}
#else
#define USE_ASYNC_LDS 0
#endif

__global__ void absmean_part(const float* __restrict__ w, long long n,
                             float* __restrict__ acc) {
    __shared__ float red[256];
    float s = 0.f;
    for (long long i = (long long)blockIdx.x * blockDim.x + threadIdx.x; i < n;
         i += (long long)gridDim.x * blockDim.x)
        s += fabsf(w[i]);
    red[threadIdx.x] = s;
    __syncthreads();
    for (int o = 128; o > 0; o >>= 1) {
        if ((int)threadIdx.x < o) red[threadIdx.x] += red[threadIdx.x + o];
        __syncthreads();
    }
    if (threadIdx.x == 0) atomicAdd(acc, red[0]);
}

__global__ void finalize_scales(float* __restrict__ stats) {
    stats[2] = fmaxf(stats[0] / (float)((size_t)HID * DIM), EPSF);
    stats[3] = fmaxf(stats[1] / (float)((size_t)DIM * HID), EPSF);
}

__global__ void wquant(const float* __restrict__ w, signed char* __restrict__ q,
                       int n, const float* __restrict__ meanPtr) {
    const float inv = 1.0f / *meanPtr;
    for (int i = blockIdx.x * blockDim.x + threadIdx.x; i < n;
         i += gridDim.x * blockDim.x) {
        float r = rintf(w[i] * inv);
        r = fminf(fmaxf(r, -1.f), 1.f);
        q[i] = (signed char)r;
    }
}

template <int W>
__global__ __launch_bounds__(256) void ln_actquant(
    const float* __restrict__ x, const float* __restrict__ gam,
    const float* __restrict__ bet, signed char* __restrict__ q,
    float* __restrict__ invs) {
    constexpr int T = 256;
    constexpr int V = W / T;
    __shared__ float red[T];
    const int tid = threadIdx.x;
    const size_t base = (size_t)blockIdx.x * W;

    float v[V];
    float s = 0.f;
#pragma unroll
    for (int i = 0; i < V; ++i) {
        v[i] = x[base + i * T + tid];
        s += v[i];
    }
    red[tid] = s;
    __syncthreads();
    for (int o = T / 2; o > 0; o >>= 1) {
        if (tid < o) red[tid] += red[tid + o];
        __syncthreads();
    }
    const float mu = red[0] / (float)W;
    __syncthreads();

    float s2 = 0.f;
#pragma unroll
    for (int i = 0; i < V; ++i) {
        const float d = v[i] - mu;
        s2 += d * d;
    }
    red[tid] = s2;
    __syncthreads();
    for (int o = T / 2; o > 0; o >>= 1) {
        if (tid < o) red[tid] += red[tid + o];
        __syncthreads();
    }
    const float rstd = rsqrtf(red[0] / (float)W + EPSF);
    __syncthreads();

    float amax = 0.f;
#pragma unroll
    for (int i = 0; i < V; ++i) {
        const int c = i * T + tid;
        v[i] = (v[i] - mu) * rstd * gam[c] + bet[c];
        amax = fmaxf(amax, fabsf(v[i]));
    }
    red[tid] = amax;
    __syncthreads();
    for (int o = T / 2; o > 0; o >>= 1) {
        if (tid < o) red[tid] = fmaxf(red[tid], red[tid + o]);
        __syncthreads();
    }
    const float am = fmaxf(red[0], EPSF);
    const float scale = 127.f / am;
    if (tid == 0) invs[blockIdx.x] = am / 127.f;
#pragma unroll
    for (int i = 0; i < V; ++i) {
        float r = rintf(v[i] * scale);
        r = fminf(fmaxf(r, -128.f), 127.f);
        q[base + i * T + tid] = (signed char)r;
    }
}

__global__ __launch_bounds__(256) void gemm_iu8(
    const signed char* __restrict__ A, const signed char* __restrict__ B,
    const float* __restrict__ rowScale, const float* __restrict__ wScalePtr,
    const float* __restrict__ bias, float* __restrict__ C, int K, int N,
    int doGelu) {
    __shared__ __align__(16) signed char lA[2][128 * 64];
    __shared__ __align__(16) signed char lB[2][128 * 64];

    const int tid  = threadIdx.x;
    const int lane = tid & 31;
    const int wid  = tid >> 5;
    const int wm   = wid & 3;
    const int wn   = wid >> 2;
    const int lm   = lane & 15;
    const int lh   = lane >> 4;
    const int mBase = blockIdx.y * 128;
    const int nBase = blockIdx.x * 128;

    v8i acc[2][4];
#pragma unroll
    for (int i = 0; i < 2; ++i)
#pragma unroll
        for (int j = 0; j < 4; ++j) acc[i][j] = (v8i)0;

    const int ldr = tid >> 1;
    const int ldc = (tid & 1) * 32;
    const signed char* gA = A + (size_t)(mBase + ldr) * K + ldc;
    const signed char* gB = B + (size_t)(nBase + ldr) * K + ldc;

    const int nk = K >> 6;

#if USE_ASYNC_LDS
    async_copy16(gA,      &lA[0][ldr * 64 + ldc]);
    async_copy16(gA + 16, &lA[0][ldr * 64 + ldc + 16]);
    async_copy16(gB,      &lB[0][ldr * 64 + ldc]);
    async_copy16(gB + 16, &lB[0][ldr * 64 + ldc + 16]);
#else
    int4 ra0, ra1, rb0, rb1;
    ra0 = *(const int4*)(gA);
    ra1 = *(const int4*)(gA + 16);
    rb0 = *(const int4*)(gB);
    rb1 = *(const int4*)(gB + 16);
    {
        int4* pa = (int4*)&lA[0][ldr * 64 + ldc];
        pa[0] = ra0; pa[1] = ra1;
        int4* pb = (int4*)&lB[0][ldr * 64 + ldc];
        pb[0] = rb0; pb[1] = rb1;
    }
#endif

    for (int k = 0; k < nk; ++k) {
        const int cur = k & 1, nxt = cur ^ 1;
        if (k + 2 < nk) {
            const size_t go2 = (size_t)(k + 2) << 6;
            __builtin_prefetch(gA + go2, 0, 0);
            __builtin_prefetch(gB + go2, 0, 0);
        }
#if USE_ASYNC_LDS
        __builtin_amdgcn_s_wait_asynccnt(0);
        __syncthreads();
        if (k + 1 < nk) {
            const size_t go = (size_t)(k + 1) << 6;
            async_copy16(gA + go,      &lA[nxt][ldr * 64 + ldc]);
            async_copy16(gA + go + 16, &lA[nxt][ldr * 64 + ldc + 16]);
            async_copy16(gB + go,      &lB[nxt][ldr * 64 + ldc]);
            async_copy16(gB + go + 16, &lB[nxt][ldr * 64 + ldc + 16]);
        }
#else
        if (k + 1 < nk) {
            const size_t go = (size_t)(k + 1) << 6;
            ra0 = *(const int4*)(gA + go);
            ra1 = *(const int4*)(gA + go + 16);
            rb0 = *(const int4*)(gB + go);
            rb1 = *(const int4*)(gB + go + 16);
        }
        __syncthreads();
        if (k + 1 < nk) {
            int4* pa = (int4*)&lA[nxt][ldr * 64 + ldc];
            pa[0] = ra0; pa[1] = ra1;
            int4* pb = (int4*)&lB[nxt][ldr * 64 + ldc];
            pb[0] = rb0; pb[1] = rb1;
        }
#endif

        v8i af[2], bf[4];
#pragma unroll
        for (int tm = 0; tm < 2; ++tm) {
            const signed char* ap = &lA[cur][(wm * 32 + tm * 16 + lm) * 64];
            const int2 t0 = *(const int2*)(ap + 0  + lh * 16);
            const int2 t1 = *(const int2*)(ap + 8  + lh * 16);
            const int2 t2 = *(const int2*)(ap + 32 + lh * 16);
            const int2 t3 = *(const int2*)(ap + 40 + lh * 16);
            v8i a;
            a[0] = t0.x; a[1] = t0.y; a[2] = t1.x; a[3] = t1.y;
            a[4] = t2.x; a[5] = t2.y; a[6] = t3.x; a[7] = t3.y;
            af[tm] = a;
        }
#pragma unroll
        for (int tn = 0; tn < 4; ++tn) {
            const signed char* bp = &lB[cur][(wn * 64 + tn * 16 + lm) * 64];
            const int4 u0 = *(const int4*)(bp + 0  + lh * 16);
            const int4 u1 = *(const int4*)(bp + 32 + lh * 16);
            v8i b;
            b[0] = u0.x; b[1] = u0.y; b[2] = u0.z; b[3] = u0.w;
            b[4] = u1.x; b[5] = u1.y; b[6] = u1.z; b[7] = u1.w;
            bf[tn] = b;
        }
#pragma unroll
        for (int tm = 0; tm < 2; ++tm)
#pragma unroll
            for (int tn = 0; tn < 4; ++tn)
                acc[tm][tn] = __builtin_amdgcn_wmma_i32_16x16x64_iu8(
                    true, af[tm], true, bf[tn], acc[tm][tn], false, false);
    }

    const float wS = *wScalePtr;
#pragma unroll
    for (int tm = 0; tm < 2; ++tm) {
#pragma unroll
        for (int tn = 0; tn < 4; ++tn) {
            const int gcol = nBase + wn * 64 + tn * 16 + lm;
            const float bv = bias[gcol];
#pragma unroll
            for (int r = 0; r < 8; ++r) {
                const int grow = mBase + wm * 32 + tm * 16 + lh * 8 + r;
                float v = (float)acc[tm][tn][r] * (rowScale[grow] * wS) + bv;
                if (doGelu)
                    v = 0.5f * v * (1.0f + erff(v * 0.70710678118654752f));
                C[(size_t)grow * N + gcol] = v;
            }
        }
    }
}

extern "C" void kernel_launch(void* const* d_in, const int* in_sizes, int n_in,
                              void* d_out, int out_size, void* d_ws,
                              size_t ws_size, hipStream_t stream) {
    const float* x    = (const float*)d_in[0];
    const float* ln1w = (const float*)d_in[1];
    const float* ln1b = (const float*)d_in[2];
    const float* w1   = (const float*)d_in[3];
    const float* b1   = (const float*)d_in[4];
    const float* ln2w = (const float*)d_in[5];
    const float* ln2b = (const float*)d_in[6];
    const float* w2   = (const float*)d_in[7];
    const float* b2   = (const float*)d_in[8];
    float* out = (float*)d_out;

    char* ws = (char*)d_ws;
    float*       stats = (float*)(ws + OFF_STATS);
    signed char* xq    = (signed char*)(ws + OFF_XQ);
    float*       sx    = (float*)(ws + OFF_SX);
    signed char* w1q   = (signed char*)(ws + OFF_W1Q);
    signed char* w2q   = (signed char*)(ws + OFF_W2Q);
    float*       h     = (float*)(ws + OFF_H);
    signed char* hq    = (signed char*)(ws + OFF_HQ);
    float*       sh    = (float*)(ws + OFF_SH);

    (void)hipMemsetAsync(stats, 0, 256, stream);

    absmean_part<<<512, 256, 0, stream>>>(w1, (long long)HID * DIM, stats + 0);
    absmean_part<<<512, 256, 0, stream>>>(w2, (long long)DIM * HID, stats + 1);
    finalize_scales<<<1, 1, 0, stream>>>(stats);
    wquant<<<1024, 256, 0, stream>>>(w1, w1q, HID * DIM, stats + 2);
    wquant<<<1024, 256, 0, stream>>>(w2, w2q, DIM * HID, stats + 3);

    ln_actquant<DIM><<<MROWS, 256, 0, stream>>>(x, ln1w, ln1b, xq, sx);

    dim3 g1(HID / 128, MROWS / 128);
    gemm_iu8<<<g1, 256, 0, stream>>>(xq, w1q, sx, stats + 2, b1, h, DIM, HID, 1);

    ln_actquant<HID><<<MROWS, 256, 0, stream>>>(h, ln2w, ln2b, hq, sh);

    dim3 g2(DIM / 128, MROWS / 128);
    gemm_iu8<<<g2, 256, 0, stream>>>(hq, w2q, sh, stats + 3, b2, out, HID, DIM, 0);
}
